// paperGIN_14199161880830
// MI455X (gfx1250) — compile-verified, matrix operand order checked by reading, not run on hardware
//
#include <hip/hip_runtime.h>
#include <hip/hip_bf16.h>

typedef __attribute__((ext_vector_type(16))) __bf16 v16bf;
typedef __attribute__((ext_vector_type(8)))  float  v8f;
typedef __attribute__((ext_vector_type(4)))  int    v4i;

#define DFEAT 128
#define AS1 __attribute__((address_space(1)))
#define AS3 __attribute__((address_space(3)))

#if defined(__gfx1250__) && \
    __has_builtin(__builtin_amdgcn_global_load_async_to_lds_b128) && \
    __has_builtin(__builtin_amdgcn_s_wait_asynccnt)
#define USE_ASYNC_LDS 1
#else
#define USE_ASYNC_LDS 0
#endif

__device__ __forceinline__ unsigned short f2bf(float f) {
    unsigned int u = __builtin_bit_cast(unsigned int, f);
    unsigned int r = u + 0x7FFFu + ((u >> 16) & 1u);   // round-to-nearest-even
    return (unsigned short)(r >> 16);
}

// ---------------------------------------------------------------------------
// WMMA GEMM:  out[M,128] = act( A[M,128](bf16) @ W[128,128] + bias )
// A is packed bf16 rows (2 halfs per dword, row stride 64 dwords).
// Wp is the weight pre-packed as pairs: Wp[n*64 + p] = (W[2p][n], W[2p+1][n]).
// Optional idx: A row for output-row r is A[idx[r]] (fused embedding gather).
// Block: 256 thr = 8 waves; wave w computes rows [blk*128+w*16, +16) x 128.
// ---------------------------------------------------------------------------
__global__ __launch_bounds__(256) void gemm_bf16_128(
    const unsigned int* __restrict__ A, const int* __restrict__ idx,
    const unsigned int* __restrict__ Wp, const float* __restrict__ bias,
    float* __restrict__ outF, unsigned short* __restrict__ outB,
    int M, int relu)
{
    __shared__ unsigned int lds[8192];                  // 32 KB packed weights

    const int wave = threadIdx.x >> 5;
    const int lane = threadIdx.x & 31;
    const int ln   = lane & 15;
    const int kg   = lane >> 4;
    const int rowbase = blockIdx.x * 128 + wave * 16;

    int ar = rowbase + ln; if (ar > M - 1) ar = M - 1;  // clamp, keep EXEC full
    const long arow = idx ? (long)idx[ar] : (long)ar;
    const unsigned int* abase = A + arow * 64;

    // ---- issue ALL A-fragment loads first (overlap with LDS weight fill) ----
    union { uint4 u[8]; v16bf v[4]; } aall;
#pragma unroll
    for (int kt = 0; kt < 4; ++kt) {
        aall.u[kt * 2]     = *(const uint4*)(abase + kt * 16 + kg * 4);
        aall.u[kt * 2 + 1] = *(const uint4*)(abase + kt * 16 + 8 + kg * 4);
    }

    // ---- stage packed weights into LDS ----
#if USE_ASYNC_LDS
    for (int j = threadIdx.x; j < 2048; j += 256)
        __builtin_amdgcn_global_load_async_to_lds_b128(
            (AS1 v4i*)(Wp + (size_t)j * 4),
            (AS3 v4i*)(lds + (size_t)j * 4), 0, 0);
    __builtin_amdgcn_s_wait_asynccnt(0);
#else
    {
        const uint4* s = (const uint4*)Wp;
        uint4* d = (uint4*)lds;
        for (int j = threadIdx.x; j < 2048; j += 256) d[j] = s[j];
    }
#endif
    __syncthreads();

    v8f acc[8] = {};

#pragma unroll
    for (int kt = 0; kt < 4; ++kt) {
#pragma unroll
        for (int nt = 0; nt < 8; ++nt) {
            // B fragment: 32x16 tile, per-lane col = nt*16+ln
            union { uint4 u[2]; v16bf v; } bfr;
            const unsigned int* bb = lds + (nt * 16 + ln) * 64 + kt * 16 + kg * 4;
            bfr.u[0] = *(const uint4*)bb;
            bfr.u[1] = *(const uint4*)(bb + 8);
            acc[nt] = __builtin_amdgcn_wmma_f32_16x16x32_bf16(
                false, aall.v[kt], false, bfr.v, (short)0, acc[nt], false, false);
        }
    }

#pragma unroll
    for (int nt = 0; nt < 8; ++nt) {
        const int col = nt * 16 + ln;
        const float bv = bias[col];
#pragma unroll
        for (int v = 0; v < 8; ++v) {
            const int row = rowbase + kg * 8 + v;       // C layout: m = kg*8+v
            if (row < M) {
                float val = acc[nt][v] + bv;
                if (relu) val = fmaxf(val, 0.0f);
                if (outF) outF[(long)row * DFEAT + col] = val;
                if (outB) outB[(long)row * DFEAT + col] = f2bf(val);
            }
        }
    }
}

// ---------------------------------------------------------------------------
// Pack 9 f32 [128,128] weight matrices into WMMA-B packed bf16 layout.
// ---------------------------------------------------------------------------
struct W9 { const float* w[9]; };

__global__ void pack_weights(W9 w9, unsigned int* __restrict__ wp)
{
    int i = blockIdx.x * blockDim.x + threadIdx.x;
    if (i >= 9 * 8192) return;
    int mat = i >> 13, r = i & 8191;
    int n = r >> 6, p = r & 63;
    const float* W = w9.w[mat];
    unsigned lo = f2bf(W[(2 * p) * DFEAT + n]);
    unsigned hi = f2bf(W[(2 * p + 1) * DFEAT + n]);
    wp[i] = lo | (hi << 16);
}

// dst[i] = packed bf16 of (a[2i](+b[2i]), a[2i+1](+b[2i+1]))
__global__ void pack_rows(const float* __restrict__ a, const float* __restrict__ b,
                          unsigned int* __restrict__ dst, long ndw)
{
    long i = (long)blockIdx.x * blockDim.x + threadIdx.x;
    if (i >= ndw) return;
    long j = i * 2;
    float lo = a[j], hi = a[j + 1];
    if (b) { lo += b[j]; hi += b[j + 1]; }
    dst[i] = (unsigned)f2bf(lo) | ((unsigned)f2bf(hi) << 16);
}

__global__ void zero_f32(float* __restrict__ p, long n)
{
    long i = (long)blockIdx.x * blockDim.x + threadIdx.x;
    if (i < n) p[i] = 0.0f;
}

// agg[dst] += h[src] : 16 threads per edge, 8 floats each
__global__ void edge_scatter(const int* __restrict__ src, const int* __restrict__ dst,
                             const float* __restrict__ h, float* __restrict__ agg, long E)
{
    long t = (long)blockIdx.x * blockDim.x + threadIdx.x;
    if (t >= E * 16) return;
    long e = t >> 4;
    int part = (int)(t & 15);
    long sb = (long)src[e] * DFEAT + part * 8;
    long db = (long)dst[e] * DFEAT + part * 8;
    float4 v0 = *(const float4*)(h + sb);
    float4 v1 = *(const float4*)(h + sb + 4);
    float vv[8] = {v0.x, v0.y, v0.z, v0.w, v1.x, v1.y, v1.z, v1.w};
#pragma unroll
    for (int i = 0; i < 8; ++i)
        __hip_atomic_fetch_add(agg + db + i, vv[i],
                               __ATOMIC_RELAXED, __HIP_MEMORY_SCOPE_AGENT);
}

// per-column partial sum / sumsq -> stats[0:128]=sum, stats[128:256]=sumsq
__global__ void bn_reduce(const float* __restrict__ y, float* __restrict__ stats, int N)
{
    int col  = threadIdx.x & 127;
    int half = threadIdx.x >> 7;
    float s = 0.0f, s2 = 0.0f;
    for (long r = blockIdx.x * 2 + half; r < N; r += (long)gridDim.x * 2) {
        float v = y[r * DFEAT + col];
        s += v; s2 += v * v;
    }
    __hip_atomic_fetch_add(stats + col,       s,  __ATOMIC_RELAXED, __HIP_MEMORY_SCOPE_AGENT);
    __hip_atomic_fetch_add(stats + 128 + col, s2, __ATOMIC_RELAXED, __HIP_MEMORY_SCOPE_AGENT);
}

__global__ void bn_finalize(const float* __restrict__ stats,
                            const float* __restrict__ g, const float* __restrict__ be,
                            float* __restrict__ sc, float* __restrict__ sh, int N)
{
    int c = threadIdx.x;
    if (c >= DFEAT) return;
    float mean = stats[c] / (float)N;
    float var  = stats[128 + c] / (float)N - mean * mean;
    float scale = g[c] * rsqrtf(var + 1e-5f);
    sc[c] = scale;
    sh[c] = be[c] - mean * scale;
}

__global__ void bn_relu(const float* __restrict__ y, const float* __restrict__ sc,
                        const float* __restrict__ sh, float* __restrict__ h, long n)
{
    long i = (long)blockIdx.x * blockDim.x + threadIdx.x;
    if (i >= n) return;
    int c = (int)(i & 127);
    h[i] = fmaxf(y[i] * sc[c] + sh[c], 0.0f);
}

__global__ void pool_add(const float* __restrict__ h, const int* __restrict__ batch,
                         float* __restrict__ pool, long n)
{
    long i = (long)blockIdx.x * blockDim.x + threadIdx.x;
    if (i >= n) return;
    long row = i >> 7;
    int  c   = (int)(i & 127);
    __hip_atomic_fetch_add(pool + (long)batch[row] * DFEAT + c, h[i],
                           __ATOMIC_RELAXED, __HIP_MEMORY_SCOPE_AGENT);
}

__global__ void final_lin(const float* __restrict__ p, const float* __restrict__ w,
                          const float* __restrict__ b, float* __restrict__ out,
                          int G, int OUT)
{
    int t = blockIdx.x * blockDim.x + threadIdx.x;
    if (t >= G * OUT) return;
    int g = t / OUT, o = t % OUT;
    float s = b[o];
    const float* pr = p + (long)g * DFEAT;
#pragma unroll 8
    for (int k = 0; k < DFEAT; ++k) s += pr[k] * w[k * OUT + o];
    out[t] = s;
}

// ---------------------------------------------------------------------------
extern "C" void kernel_launch(void* const* d_in, const int* in_sizes, int n_in,
                              void* d_out, int out_size, void* d_ws, size_t ws_size,
                              hipStream_t stream)
{
    const int*   x     = (const int*)d_in[0];
    const int*   ei    = (const int*)d_in[1];
    const int*   batch = (const int*)d_in[2];
    const float* emb   = (const float*)d_in[3];
    const float* iw1   = (const float*)d_in[4];
    const float* ib1   = (const float*)d_in[5];
    const float* iw2   = (const float*)d_in[6];
    const float* ib2   = (const float*)d_in[7];
    const float* f1w   = (const float*)d_in[8];
    const float* f1b   = (const float*)d_in[9];
    const float* f2w   = (const float*)d_in[10];
    const float* f2b   = (const float*)d_in[11];

    const int N = in_sizes[0];
    const long E = in_sizes[1] / 2;
    const int VOCAB = in_sizes[3] / DFEAT;
    const int OUT = in_sizes[11];
    const int G = out_size / OUT;

    // conv layer params: index 12 + 6*l : {w1, b1, w2, b2, g, be}
    const float* cw1[3]; const float* cb1[3]; const float* cw2[3];
    const float* cb2[3]; const float* cg[3];  const float* cbe[3];
    for (int l = 0; l < 3; ++l) {
        cw1[l] = (const float*)d_in[12 + 6 * l + 0];
        cb1[l] = (const float*)d_in[12 + 6 * l + 1];
        cw2[l] = (const float*)d_in[12 + 6 * l + 2];
        cb2[l] = (const float*)d_in[12 + 6 * l + 3];
        cg[l]  = (const float*)d_in[12 + 6 * l + 4];
        cbe[l] = (const float*)d_in[12 + 6 * l + 5];
    }

    // ---- carve workspace ----
    char* base = (char*)d_ws;
    size_t off = 0;
    auto carve = [&](size_t bytes) -> void* {
        void* p = base + off;
        off = (off + bytes + 255) & ~(size_t)255;
        return p;
    };
    unsigned int*  wpAll = (unsigned int*) carve((size_t)9 * 8192 * 4);
    unsigned int*  embA  = (unsigned int*) carve((size_t)VOCAB * 64 * 4);
    unsigned int*  aB    = (unsigned int*) carve((size_t)N * 64 * 4);     // bf16 A buf
    unsigned int*  tB    = (unsigned int*) carve((size_t)N * 64 * 4);     // bf16 hidden
    float*         hF    = (float*)        carve((size_t)N * DFEAT * 4);
    float*         yF    = (float*)        carve((size_t)N * DFEAT * 4);
    float*         agg   = (float*)        carve((size_t)N * DFEAT * 4);
    float*         stats = (float*)        carve(256 * 4);
    float*         sc    = (float*)        carve(128 * 4);
    float*         sh    = (float*)        carve(128 * 4);
    float*         pool  = (float*)        carve((size_t)G * DFEAT * 4);
    unsigned int*  poolB = (unsigned int*) carve((size_t)G * 64 * 4);
    float*         pF    = (float*)        carve((size_t)G * DFEAT * 4);

    // ---- pack weights (bf16, WMMA-B layout) ----
    W9 w9;
    w9.w[0] = iw1; w9.w[1] = iw2;
    w9.w[2] = cw1[0]; w9.w[3] = cw2[0];
    w9.w[4] = cw1[1]; w9.w[5] = cw2[1];
    w9.w[6] = cw1[2]; w9.w[7] = cw2[2];
    w9.w[8] = f1w;
    pack_weights<<<(9 * 8192 + 255) / 256, 256, 0, stream>>>(w9, wpAll);

    // embedding table -> packed bf16 rows
    {
        long ndw = (long)VOCAB * 64;
        pack_rows<<<(int)((ndw + 255) / 256), 256, 0, stream>>>(emb, nullptr, embA, ndw);
    }

    auto gemm = [&](const unsigned int* A, const int* idx, int mat, const float* bias,
                    float* outF, unsigned int* outB, int M, int relu) {
        gemm_bf16_128<<<(M + 127) / 128, 256, 0, stream>>>(
            A, idx, wpAll + (size_t)mat * 8192, bias,
            outF, (unsigned short*)outB, M, relu);
    };

    const long nElem = (long)N * DFEAT;
    const long nDw   = (long)N * 64;

    // ---- initial MLP (embedding gather fused into first GEMM) ----
    gemm(embA, x, 0, ib1, nullptr, tB, N, 1);      // t = relu(emb[x] @ iw1 + ib1)
    gemm(tB, nullptr, 1, ib2, hF, nullptr, N, 0);  // h = t @ iw2 + ib2

    // ---- 3 GIN conv layers ----
    for (int l = 0; l < 3; ++l) {
        zero_f32<<<(int)((nElem + 255) / 256), 256, 0, stream>>>(agg, nElem);
        edge_scatter<<<(int)((E * 16 + 255) / 256), 256, 0, stream>>>(
            ei, ei + E, hF, agg, E);
        // a = bf16(h + agg)
        pack_rows<<<(int)((nDw + 255) / 256), 256, 0, stream>>>(hF, agg, aB, nDw);
        gemm(aB, nullptr, 2 + 2 * l, cb1[l], nullptr, tB, N, 1);
        gemm(tB, nullptr, 3 + 2 * l, cb2[l], yF, nullptr, N, 0);
        zero_f32<<<1, 256, 0, stream>>>(stats, 256);
        bn_reduce<<<256, 256, 0, stream>>>(yF, stats, N);
        bn_finalize<<<1, 128, 0, stream>>>(stats, cg[l], cbe[l], sc, sh, N);
        bn_relu<<<(int)((nElem + 255) / 256), 256, 0, stream>>>(yF, sc, sh, hF, nElem);
    }

    // ---- global add pool + head ----
    {
        long pn = (long)G * DFEAT;
        zero_f32<<<(int)((pn + 255) / 256), 256, 0, stream>>>(pool, pn);
        pool_add<<<(int)((nElem + 255) / 256), 256, 0, stream>>>(hF, batch, pool, nElem);
        long pdw = (long)G * 64;
        pack_rows<<<(int)((pdw + 255) / 256), 256, 0, stream>>>(pool, nullptr, poolB, pdw);
        gemm(poolB, nullptr, 8, f1b, pF, nullptr, G, 1);  // p = relu(pool @ f1w + f1b)
        final_lin<<<(G * OUT + 255) / 256, 256, 0, stream>>>(
            pF, f2w, f2b, (float*)d_out, G, OUT);
    }
}
